// Model_16217796510271
// MI455X (gfx1250) — matrix operand order repaired and checked by reading, not compiled, not run on hardware
//
#include <hip/hip_runtime.h>
#include <hip/hip_bf16.h>

typedef __bf16 v16bf __attribute__((ext_vector_type(16)));
typedef __bf16 v8bf  __attribute__((ext_vector_type(8)));
typedef float  v8f   __attribute__((ext_vector_type(8)));

#define NUM_J 17
#define DIM   256
#define BT    15552                 // 64*243
#define NTOK  (BT*NUM_J)            // 264384 (divisible by 16)
#define XROW  264                   // padded bf16 LDS row stride (banks 0,4,8,... across M lanes)

// norm_adj = D^{-1/2} A D^{-1/2}, <=2 nonzeros per row; compile-time constants.
#define RT2 0.70710678118654752f
__constant__ int   c_nbr0[NUM_J] = {1,2,3,2,5,6,5,0,7,8,9,12,13,12,15,16,15};
__constant__ int   c_nbr1[NUM_J] = {7,0,1,0,0,4,0,8,9,10,0,8,11,0,8,14,0};
__constant__ float c_w0  [NUM_J] = {0.5f,0.5f,RT2,RT2,0.5f,RT2,RT2,0.5f,0.5f,0.5f,RT2,0.5f,RT2,RT2,0.5f,RT2,RT2};
__constant__ float c_w1  [NUM_J] = {0.5f,0.5f,0.5f,0.f,0.5f,0.5f,0.f,0.5f,0.5f,RT2,0.f,0.5f,0.5f,0.f,0.5f,0.5f,0.f};

__device__ __forceinline__ unsigned short f2bf(float f) {
    unsigned u = __float_as_uint(f);
    u += 0x7fffu + ((u >> 16) & 1u);          // round-to-nearest-even
    return (unsigned short)(u >> 16);
}

// A fragment (16x32 bf16, M x K) from an LDS tile with row stride XROW.
// lane: m = lane&15, half = lane>>4. element i: k = half*16 + i.
__device__ __forceinline__ v16bf load_afrag(const unsigned short* tile, int m, int k0, int half) {
    const __bf16* row = (const __bf16*)tile + m * XROW;
    v8bf lo = *(const v8bf*)(row + k0 + half * 16);
    v8bf hi = *(const v8bf*)(row + k0 + half * 16 + 8);
    return __builtin_shufflevector(lo, hi, 0,1,2,3,4,5,6,7,8,9,10,11,12,13,14,15);
}

// ---------------------------------------------------------------- prep: f32 -> bf16 weights
__global__ void wconvert(const float* __restrict__ Vw, const float* __restrict__ Uw,
                         const float* __restrict__ Aw, unsigned short* __restrict__ dst) {
    int i = blockIdx.x * 256 + threadIdx.x;
    if (i < 65536)        dst[i] = f2bf(Vw[i]);
    else if (i < 131072)  dst[i] = f2bf(Uw[i - 65536]);
    else if (i < 147456)  dst[i] = f2bf(Aw[i - 131072]);
}

// ---------------------------------------------------------------- main: U/V GEMM + adj mix + BN partials
__global__ __launch_bounds__(256)
void gcn_gemm_mix(const float* __restrict__ x,
                  const unsigned short* __restrict__ Wv, const unsigned short* __restrict__ Wu,
                  const float* __restrict__ Vb, const float* __restrict__ Ub,
                  float* __restrict__ y_out, float* __restrict__ partials) {
    __shared__ __align__(16) unsigned short xt[32 * XROW];   // rows 0..16 valid; rows 17..31 unused garbage
    __shared__ __align__(16) float vxs[NUM_J * DIM];
    __shared__ __align__(16) float yts[NUM_J * DIM];

    const int bt   = blockIdx.x;
    const int tid  = threadIdx.x;
    const int wave = tid >> 5, lane = tid & 31, half = lane >> 4, l16 = lane & 15;

    // load x slab (17x256 f32) -> bf16 LDS tile
    const float* xb = x + (size_t)bt * NUM_J * DIM;
    for (int i = tid; i < NUM_J * DIM / 4; i += 256) {
        int c = i * 4, row = c >> 8, col = c & 255;
        float4 v = *(const float4*)(xb + c);
        unsigned short* p = xt + row * XROW + col;
        p[0] = f2bf(v.x); p[1] = f2bf(v.y); p[2] = f2bf(v.z); p[3] = f2bf(v.w);
    }
    __syncthreads();

    const bool isV = (wave < 4);
    const unsigned short* W = isV ? Wv : Wu;
    const int n0 = (wave & 3) * 64;

    v8f acc[2][4];
#pragma unroll
    for (int mt = 0; mt < 2; ++mt)
#pragma unroll
        for (int nt = 0; nt < 4; ++nt)
            acc[mt][nt] = (v8f){0.f,0.f,0.f,0.f,0.f,0.f,0.f,0.f};

#pragma unroll
    for (int k0 = 0; k0 < DIM; k0 += 32) {
        v16bf a0 = load_afrag(xt, l16,      k0, half);
        v16bf a1 = load_afrag(xt, 16 + l16, k0, half);
#pragma unroll
        for (int nt = 0; nt < 4; ++nt) {
            int n = n0 + nt * 16 + l16;
            v16bf b = *(const v16bf*)((const __bf16*)W + n * DIM + k0 + half * 16);
            acc[0][nt] = __builtin_amdgcn_wmma_f32_16x16x32_bf16(false, a0, false, b, (short)0, acc[0][nt], false, false);
            acc[1][nt] = __builtin_amdgcn_wmma_f32_16x16x32_bf16(false, a1, false, b, (short)0, acc[1][nt], false, false);
        }
    }

    const float* bias = isV ? Vb : Ub;
    if (isV) {            // waves 0-3: vx (+bias) -> LDS
#pragma unroll
        for (int mt = 0; mt < 2; ++mt)
#pragma unroll
            for (int nt = 0; nt < 4; ++nt) {
                int n = n0 + nt * 16 + l16;
                float bv = bias[n];
#pragma unroll
                for (int r = 0; r < 8; ++r) {
                    int m = mt * 16 + r + half * 8;
                    if (m < NUM_J) vxs[m * DIM + n] = acc[mt][nt][r] + bv;
                }
            }
    }
    __syncthreads();
    if (!isV) {           // waves 4-7: y = adj*vx + ux
#pragma unroll
        for (int mt = 0; mt < 2; ++mt)
#pragma unroll
            for (int nt = 0; nt < 4; ++nt) {
                int n = n0 + nt * 16 + l16;
                float bv = bias[n];
#pragma unroll
                for (int r = 0; r < 8; ++r) {
                    int m = mt * 16 + r + half * 8;
                    if (m < NUM_J) {
                        float val = acc[mt][nt][r] + bv
                                  + c_w0[m] * vxs[c_nbr0[m] * DIM + n]
                                  + c_w1[m] * vxs[c_nbr1[m] * DIM + n];
                        yts[m * DIM + n] = val;
                    }
                }
            }
    }
    __syncthreads();

    // coalesced store of y (17x256 contiguous per (b,t))
    float* yg = y_out + (size_t)bt * NUM_J * DIM;
    for (int i = tid; i < NUM_J * DIM / 4; i += 256)
        ((float4*)yg)[i] = ((const float4*)yts)[i];

    // per-joint BN partials (deterministic shuffle-tree over 8 lanes)
    if (tid < NUM_J * 8) {
        int j = tid >> 3, l8 = tid & 7;
        float s = 0.f, s2 = 0.f;
        const float* yr = yts + j * DIM + l8 * 32;
#pragma unroll
        for (int c = 0; c < 32; ++c) { float v = yr[c]; s += v; s2 += v * v; }
        s  += __shfl_xor(s, 4, 8);  s2 += __shfl_xor(s2, 4, 8);
        s  += __shfl_xor(s, 2, 8);  s2 += __shfl_xor(s2, 2, 8);
        s  += __shfl_xor(s, 1, 8);  s2 += __shfl_xor(s2, 1, 8);
        if (l8 == 0) { partials[bt * 34 + j * 2] = s; partials[bt * 34 + j * 2 + 1] = s2; }
    }
}

// ---------------------------------------------------------------- BN stats -> (scale, shift) per joint
__global__ void bn_finalize(const float* __restrict__ partials,
                            const float* __restrict__ bn_w, const float* __restrict__ bn_b,
                            float2* __restrict__ coef) {
    int j = blockIdx.x, tid = threadIdx.x;
    double s = 0.0, s2 = 0.0;
    for (int i = tid; i < BT; i += 256) {
        s  += (double)partials[i * 34 + j * 2];
        s2 += (double)partials[i * 34 + j * 2 + 1];
    }
    __shared__ double sd[256], sq[256];
    sd[tid] = s; sq[tid] = s2;
    __syncthreads();
    for (int off = 128; off > 0; off >>= 1) {
        if (tid < off) { sd[tid] += sd[tid + off]; sq[tid] += sq[tid + off]; }
        __syncthreads();
    }
    if (tid == 0) {
        double N = (double)BT * (double)DIM;
        double mean = sd[0] / N;
        double var  = sq[0] / N - mean * mean;
        float inv = (float)(1.0 / sqrt(var + 1e-5));
        float sc = bn_w[j] * inv;
        coef[j] = make_float2(sc, bn_b[j] - (float)mean * sc);
    }
}

// ---------------------------------------------------------------- relu(x + y_bn), attention gate, output
__global__ __launch_bounds__(128)
void out_att(const float* __restrict__ x, const float* __restrict__ y,
             const float2* __restrict__ coef, const unsigned short* __restrict__ Aw,
             const float* __restrict__ ab1, const float* __restrict__ aw2,
             const float* __restrict__ ab2, float* __restrict__ out) {
    __shared__ __align__(16) float outs[16 * DIM];
    __shared__ __align__(16) unsigned short ob[16 * XROW];
    __shared__ float gpart[4][16];
    __shared__ float gate[16];

    const int tid = threadIdx.x, wave = tid >> 5, lane = tid & 31, half = lane >> 4, l16 = lane & 15;
    const int tok0 = blockIdx.x * 16;
    const size_t base = (size_t)blockIdx.x * 16 * DIM;

    for (int i = tid; i < 16 * DIM / 4; i += 128) {
        int c = i * 4, row = c >> 8, col = c & 255;
        float2 cf = coef[(tok0 + row) % NUM_J];
        float4 xv = *(const float4*)(x + base + c);
        float4 yv = *(const float4*)(y + base + c);
        float4 o;
        o.x = fmaxf(xv.x + yv.x * cf.x + cf.y, 0.f);
        o.y = fmaxf(xv.y + yv.y * cf.x + cf.y, 0.f);
        o.z = fmaxf(xv.z + yv.z * cf.x + cf.y, 0.f);
        o.w = fmaxf(xv.w + yv.w * cf.x + cf.y, 0.f);
        ((float4*)outs)[i] = o;
        unsigned short* p = ob + row * XROW + col;
        p[0] = f2bf(o.x); p[1] = f2bf(o.y); p[2] = f2bf(o.z); p[3] = f2bf(o.w);
    }
    __syncthreads();

    // 16x256 @ 256x64 attention GEMM: wave w owns N-tile w
    v8f acc = (v8f){0.f,0.f,0.f,0.f,0.f,0.f,0.f,0.f};
    const int n = wave * 16 + l16;
#pragma unroll
    for (int k0 = 0; k0 < DIM; k0 += 32) {
        v16bf a = load_afrag(ob, l16, k0, half);
        v16bf b = *(const v16bf*)((const __bf16*)Aw + n * DIM + k0 + half * 16);
        acc = __builtin_amdgcn_wmma_f32_16x16x32_bf16(false, a, false, b, (short)0, acc, false, false);
    }

    const float b1 = ab1[n], w2 = aw2[n];
#pragma unroll
    for (int r = 0; r < 8; ++r) {
        float v = fmaxf(acc[r] + b1, 0.f) * w2;     // relu(h)*w2, reduce over this wave's 16 n's
        v += __shfl_xor(v, 8, 16);
        v += __shfl_xor(v, 4, 16);
        v += __shfl_xor(v, 2, 16);
        v += __shfl_xor(v, 1, 16);
        if (l16 == 0) gpart[wave][r + half * 8] = v;
    }
    __syncthreads();
    if (tid < 16) {
        float g = gpart[0][tid] + gpart[1][tid] + gpart[2][tid] + gpart[3][tid] + ab2[0];
        gate[tid] = 1.f / (1.f + expf(-g));
    }
    __syncthreads();

    for (int i = tid; i < 16 * DIM / 4; i += 128) {
        float g = gate[i >> 6];
        float4 o = ((const float4*)outs)[i];
        o.x *= g; o.y *= g; o.z *= g; o.w *= g;
        ((float4*)(out + base))[i] = o;
    }
}

// ---------------------------------------------------------------- launcher
extern "C" void kernel_launch(void* const* d_in, const int* in_sizes, int n_in,
                              void* d_out, int out_size, void* d_ws, size_t ws_size,
                              hipStream_t stream) {
    const float* x   = (const float*)d_in[0];
    const float* Uw  = (const float*)d_in[1];
    const float* Ub  = (const float*)d_in[2];
    const float* Vw  = (const float*)d_in[3];
    const float* Vb  = (const float*)d_in[4];
    const float* bnw = (const float*)d_in[5];
    const float* bnb = (const float*)d_in[6];
    const float* aw1 = (const float*)d_in[7];
    const float* ab1 = (const float*)d_in[8];
    const float* aw2 = (const float*)d_in[9];
    const float* ab2 = (const float*)d_in[10];
    float* out = (float*)d_out;

    // workspace layout: y (NTOK*DIM f32) | partials (BT*34 f32) | coef (32 float2) | bf16 weights
    float* ybuf     = (float*)d_ws;
    float* partials = ybuf + (size_t)NTOK * DIM;
    float2* coef    = (float2*)(partials + (size_t)BT * 34);
    unsigned short* wbf = (unsigned short*)(coef + 32);
    unsigned short* Wv = wbf;
    unsigned short* Wu = wbf + 65536;
    unsigned short* Wa = wbf + 131072;

    wconvert<<<576, 256, 0, stream>>>(Vw, Uw, aw1, wbf);
    gcn_gemm_mix<<<BT, 256, 0, stream>>>(x, Wv, Wu, Vb, Ub, ybuf, partials);
    bn_finalize<<<NUM_J, 256, 0, stream>>>(partials, bnw, bnb, coef);
    out_att<<<NTOK / 16, 128, 0, stream>>>(x, ybuf, coef, Wa, ab1, aw2, ab2, out);
}
